// REN_79053168050350
// MI455X (gfx1250) — matrix operand order repaired and checked by reading, not compiled, not run on hardware
//
#include <hip/hip_runtime.h>

// MI455X (gfx1250) CDNA5 implementation.
// Sequential recurrence st@U ([20,256]x[256,256] per step, 172 GFLOP total):
// one workgroup per batch chain; U^T in per-wave VGPRs (128 regs, reused
// 1024x); per-step matmul = v_wmma_f32_16x16x32_bf16 (f32 accumulate);
// state kept three ways: f32 LDS (gate dots), bf16 LDS (WMMA A-fragments,
// maintained incrementally by the update lanes), f32 registers (update chain).
// e/m streamed with double-buffered GLOBAL_LOAD_ASYNC_TO_LDS_B128 (ASYNCcnt).
// Only 2 workgroup barriers per timestep.

constexpr int kD  = 256;   // feature dim
constexpr int kNB = 20;    // slots
constexpr int kL  = 1024;  // timesteps
constexpr int kB  = 64;    // batch
constexpr int kCH = 16;    // timestep chunk staged per outer iteration
constexpr int kThreads = 512;  // 16 wave32s

typedef __attribute__((ext_vector_type(16))) __bf16 bf16x16;
typedef __attribute__((ext_vector_type(8)))  float  f32x8;

__device__ __forceinline__ unsigned short f2bf(float f) {
    unsigned u = __float_as_uint(f);
    unsigned r = u + 0x7FFFu + ((u >> 16) & 1u);   // round-to-nearest-even
    return (unsigned short)(r >> 16);
}
__device__ __forceinline__ unsigned pack2(float a, float b) {
    return (unsigned)f2bf(a) | ((unsigned)f2bf(b) << 16);
}

// A-fragment (16x32 bf16, M x K): per lane 16 contiguous bf16
__device__ __forceinline__ bf16x16 ldfragA(const unsigned short* p) {
    union { uint4 u[2]; bf16x16 v; } r;
    r.u[0] = *reinterpret_cast<const uint4*>(p);
    r.u[1] = *reinterpret_cast<const uint4*>(p + 8);
    return r.v;
}
// B-fragment (32x16 bf16, K x N), column-major storage: 16 contiguous bf16
__device__ __forceinline__ bf16x16 ldfragB(const unsigned short* p) {
    union { uint4 u[2]; bf16x16 v; } r;
    r.u[0] = *reinterpret_cast<const uint4*>(p);
    r.u[1] = *reinterpret_cast<const uint4*>(p + 8);
    return r.v;
}
__device__ __forceinline__ f32x8 wmma_bf16(bf16x16 a, bf16x16 b, f32x8 c) {
    return __builtin_amdgcn_wmma_f32_16x16x32_bf16(false, a, false, b,
                                                   (short)0, c, false, false);
}

// CDNA5 async global->LDS copy (ASYNCcnt-tracked), §10 "Async Memory".
__device__ __forceinline__ void async_b128(unsigned lds_off, const void* g) {
    asm volatile("global_load_async_to_lds_b128 %0, %1, off"
                 :: "v"(lds_off), "v"(g) : "memory");
}
__device__ __forceinline__ void wait_async0() {
    asm volatile("s_wait_asynccnt 0x0" ::: "memory");
}

// ---- precompute: U,W -> transposed bf16 (UT[n][k] = bf16(U[k][n])) ----
__global__ __launch_bounds__(256)
void k_prep(const float* __restrict__ U, const float* __restrict__ W,
            unsigned short* __restrict__ UT, unsigned short* __restrict__ WT) {
    int n = blockIdx.x, k = threadIdx.x;
    UT[n * kD + k] = f2bf(U[k * kD + n]);
    WT[n * kD + k] = f2bf(W[k * kD + n]);
}

// ---- precompute: kvb[j][d] = U_bias[d] + sum_k keys[j][k] * V[k][d] ----
__global__ __launch_bounds__(256)
void k_kvb(const float* __restrict__ keys, const float* __restrict__ V,
           const float* __restrict__ Ub, float* __restrict__ kvb) {
    int j = blockIdx.x, d = threadIdx.x;
    float s = Ub[d];
    for (int k = 0; k < kD; ++k) s += keys[j * kD + k] * V[k * kD + d];
    kvb[j * kD + d] = s;
}

// ---- dynamic LDS layout (bytes) ----
constexpr unsigned SM_STBF  = 0;        // stbf  : 32*256 bf16   = 16384 (rows 20..31 zero)
constexpr unsigned SM_STF   = 16384;    // stf   : 20*256 f32    = 20480
constexpr unsigned SM_KEYS  = 36864;    // keysf : 20*256 f32    = 20480
constexpr unsigned SM_KVB   = 57344;    // kvb   : 20*256 f32    = 20480
constexpr unsigned SM_EBF   = 77824;    // ebf   : 16*256 bf16   = 8192
constexpr unsigned SM_EWCH  = 86016;    // eWch  : 16*256 f32    = 16384
constexpr unsigned SM_VML   = 102400;   // vml   : 256 f32       = 1024
constexpr unsigned SM_MVCH  = 103424;   // mvch  : 16 f32        = 64
constexpr unsigned SM_GATE  = 103488;   // gate  : 32 f32        = 128
constexpr unsigned SM_GPART = 103616;   // gpart : 512 f32       = 2048 (mv partials)
constexpr unsigned SM_ECH   = 105664;   // ech   : 2x(16*256) f32 = 32768 (double buffer)
constexpr unsigned SM_MCH   = 138432;   // mch   : 2x(16*256) f32 = 32768 (double buffer)
constexpr unsigned SM_TOTAL = 171200;   // ~167 KB of 320 KB WGP LDS

__global__ __launch_bounds__(kThreads)
void k_scan(const float* __restrict__ e, const float* __restrict__ m,
            const float* __restrict__ state0, const float* __restrict__ keys,
            const float* __restrict__ Vm, const float* __restrict__ prelu,
            const unsigned short* __restrict__ UTg,
            const unsigned short* __restrict__ WTg,
            const float* __restrict__ kvbg, float* __restrict__ out) {
    extern __shared__ char smem[];
    unsigned short* stbf = (unsigned short*)(smem + SM_STBF);
    float* stf   = (float*)(smem + SM_STF);
    float* keysf = (float*)(smem + SM_KEYS);
    float* kvb   = (float*)(smem + SM_KVB);
    unsigned short* ebf = (unsigned short*)(smem + SM_EBF);
    float* eWch  = (float*)(smem + SM_EWCH);
    float* vml   = (float*)(smem + SM_VML);
    float* mvch  = (float*)(smem + SM_MVCH);
    float* gate  = (float*)(smem + SM_GATE);
    float* gpart = (float*)(smem + SM_GPART);
    float* ech   = (float*)(smem + SM_ECH);
    float* mch   = (float*)(smem + SM_MCH);

    const int tid  = threadIdx.x;
    const int b    = blockIdx.x;
    const int lane = tid & 31;
    const int wv   = tid >> 5;        // 0..15
    const int half = lane >> 4;       // 0/1
    const int lr   = lane & 15;
    const int mt   = wv >> 3;         // M-tile 0/1 (rows mt*16..mt*16+15)
    const int nb2  = (wv & 7) * 2;    // 2 N-tiles per wave
    const int rbase = mt * 16 + half * 8;   // first C-fragment row for this lane
    const float pa = prelu[0];

    // async prefetch of one 16-timestep chunk of e and m into LDS buffer `buf`
    auto issue_prefetch = [&](int buf, int c) {
        const char* eg = (const char*)(e + (size_t)b * kL * kD + (size_t)c * kCH * kD);
        const char* mg = (const char*)(m + (size_t)b * kL * kD + (size_t)c * kCH * kD);
        unsigned ebase = (unsigned)(size_t)(const void*)(ech + buf * 4096);
        unsigned mbase = (unsigned)(size_t)(const void*)(mch + buf * 4096);
        for (int i = tid; i < 1024; i += kThreads) {        // 16KB each, b128 granules
            async_b128(ebase + (unsigned)i * 16u, eg + (size_t)i * 16);
            async_b128(mbase + (unsigned)i * 16u, mg + (size_t)i * 16);
        }
    };

    issue_prefetch(0, 0);   // chunk 0 in flight while we stage constants

    // ---- per-wave register-resident U^T B-fragments (2 N-tiles x 8 K-steps) ----
    bf16x16 bU[2][8];
    #pragma unroll
    for (int j = 0; j < 2; ++j)
        #pragma unroll
        for (int k = 0; k < 8; ++k)
            bU[j][k] = ldfragB(UTg + ((nb2 + j) * 16 + lr) * kD + k * 32 + half * 16);

    // ---- register-resident state tile owned by this lane (C-fragment slots) ----
    float st_reg[2][8];
    #pragma unroll
    for (int j = 0; j < 2; ++j)
        #pragma unroll
        for (int i = 0; i < 8; ++i) {
            int row = rbase + i, col = (nb2 + j) * 16 + lr;
            st_reg[j][i] = (row < kNB) ? state0[row * kD + col] : 0.f;
        }

    // ---- one-time init: state (f32 + bf16), keys, kvb, V_m into LDS ----
    {
        const float4* s4 = (const float4*)state0;
        const float4* k4 = (const float4*)keys;
        const float4* v4 = (const float4*)kvbg;
        float4* sd = (float4*)stf;
        float4* kd = (float4*)keysf;
        float4* vd = (float4*)kvb;
        uint2* sb2 = (uint2*)stbf;
        for (int idx = tid; idx < 1280; idx += kThreads) {   // 5120 f32 each
            float4 f = s4[idx];
            sd[idx] = f; kd[idx] = k4[idx]; vd[idx] = v4[idx];
            uint2 p; p.x = pack2(f.x, f.y); p.y = pack2(f.z, f.w);
            sb2[idx] = p;                                    // bf16 state rows 0..19
        }
        if (tid < kD) vml[tid] = Vm[tid];
        // zero pad rows 20..31 of stbf (A-matrix M padded to 32)
        unsigned* sb = (unsigned*)stbf;
        for (int idx = tid; idx < 1536; idx += kThreads) sb[2560 + idx] = 0u;
    }
    __syncthreads();

    for (int c = 0; c < kL / kCH; ++c) {
        const int cur = c & 1;
        wait_async0();          // this wave's async chunk-c loads complete
        __syncthreads();        // ...and everyone else's

        // ---- Phase A: convert e chunk to bf16, m.V_m partial dots ----
        {
            const float4* ed = (const float4*)(ech + cur * 4096);
            uint2* eb2 = (uint2*)ebf;
            for (int idx = tid; idx < 1024; idx += kThreads) {   // 16*256 f32
                float4 f = ed[idx];
                uint2 p; p.x = pack2(f.x, f.y); p.y = pack2(f.z, f.w);
                eb2[idx] = p;
            }
            int j = tid >> 5, ii = tid & 31;    // 16 rows x 32 partials
            const float* mrow = mch + cur * 4096 + j * kD + ii * 8;
            const float* vr = vml + ii * 8;
            float s = 0.f;
            #pragma unroll
            for (int q = 0; q < 8; ++q) s += mrow[q] * vr[q];
            gpart[j * 32 + ii] = s;
        }
        if (c + 1 < kL / kCH) issue_prefetch(cur ^ 1, c + 1);   // overlap next chunk
        __syncthreads();

        // ---- Phase B: eW chunk = e_chunk @ W via WMMA; reduce mv ----
        {
            if (tid < kCH) {
                float s = 0.f;
                #pragma unroll
                for (int q = 0; q < 32; ++q) s += gpart[tid * 32 + q];
                mvch[tid] = s;
            }
            f32x8 acc = {};
            const int nt = wv;   // one N-tile per wave
            #pragma unroll
            for (int k = 0; k < 8; ++k) {
                bf16x16 af = ldfragA(ebf + lr * kD + k * 32 + half * 16);
                bf16x16 bf = ldfragB(WTg + (nt * 16 + lr) * kD + k * 32 + half * 16);
                acc = wmma_bf16(af, bf, acc);
            }
            #pragma unroll
            for (int i = 0; i < 8; ++i) {
                int row = half * 8 + i;                 // timestep in chunk
                eWch[row * kD + nt * 16 + lr] = acc[i];
            }
        }
        __syncthreads();

        // ---- sequential recurrence over the 16 staged timesteps ----
        for (int t = 0; t < kCH; ++t) {
            // Phase D: candidate matmul st@U (WMMA, U in VGPRs)
            //          + gate dot with intra-wave butterfly reduction
            f32x8 acc[2] = {};
            {
                #pragma unroll
                for (int k = 0; k < 8; ++k) {
                    bf16x16 af = ldfragA(stbf + (mt * 16 + lr) * kD + k * 32 + half * 16);
                    acc[0] = wmma_bf16(af, bU[0][k], acc[0]);
                    acc[1] = wmma_bf16(af, bU[1][k], acc[1]);
                }
                // gate_j = sigmoid((keys_j + st_j).e_t + mv_t), rows 0..19
                int j = tid >> 4, ii = tid & 15;
                float s = 0.f;
                if (j < kNB) {
                    const float* kr = keysf + j * kD + ii * 16;
                    const float* sr = stf   + j * kD + ii * 16;
                    const float* er = ech + cur * 4096 + t * kD + ii * 16;
                    #pragma unroll
                    for (int q = 0; q < 16; ++q) s += (kr[q] + sr[q]) * er[q];
                }
                s += __shfl_xor(s, 1, 16);
                s += __shfl_xor(s, 2, 16);
                s += __shfl_xor(s, 4, 16);
                s += __shfl_xor(s, 8, 16);
                if (j < kNB && ii == 0)
                    gate[j] = 1.f / (1.f + __expf(-(s + mvch[t])));
            }
            __syncthreads();
            // Phase F: st += gate * PReLU(st@U + kvb + eW_t); refresh f32+bf16 LDS
            {
                const float4 g0 = *(const float4*)(gate + rbase);
                const float4 g1 = *(const float4*)(gate + rbase + 4);
                const float gv[8] = {g0.x, g0.y, g0.z, g0.w, g1.x, g1.y, g1.z, g1.w};
                #pragma unroll
                for (int j = 0; j < 2; ++j) {
                    const int col = (nb2 + j) * 16 + lr;
                    const float ew = eWch[t * kD + col];
                    #pragma unroll
                    for (int i = 0; i < 8; ++i) {
                        const int row = rbase + i;          // C layout §7.12.2
                        if (row < kNB) {
                            float pre  = acc[j][i] + kvb[row * kD + col] + ew;
                            float cand = pre >= 0.f ? pre : pa * pre;
                            float ns   = st_reg[j][i] + gv[i] * cand;
                            st_reg[j][i] = ns;
                            stf[row * kD + col]  = ns;         // f32 for gate dots
                            stbf[row * kD + col] = f2bf(ns);   // bf16 for A-frags
                        }
                    }
                }
            }
            __syncthreads();
        }
    }
    // ---- write final state [NB, D] for this batch ----
    {
        float4* o4 = (float4*)(out + (size_t)b * kNB * kD);
        const float4* s4 = (const float4*)stf;
        for (int idx = tid; idx < 1280; idx += kThreads) o4[idx] = s4[idx];
    }
}

extern "C" void kernel_launch(void* const* d_in, const int* in_sizes, int n_in,
                              void* d_out, int out_size, void* d_ws, size_t ws_size,
                              hipStream_t stream) {
    const float* e    = (const float*)d_in[0];
    const float* m    = (const float*)d_in[1];
    const float* st0  = (const float*)d_in[2];
    const float* keys = (const float*)d_in[3];
    const float* Vm   = (const float*)d_in[4];
    const float* U    = (const float*)d_in[5];
    const float* Ub   = (const float*)d_in[6];
    const float* V    = (const float*)d_in[7];
    const float* W    = (const float*)d_in[8];
    const float* pa   = (const float*)d_in[9];

    char* ws = (char*)d_ws;
    unsigned short* UT = (unsigned short*)(ws);             // 131072 B
    unsigned short* WT = (unsigned short*)(ws + 131072);    // 131072 B
    float* kvb         = (float*)(ws + 262144);             // 20480 B

    k_prep<<<kD, 256, 0, stream>>>(U, W, UT, WT);
    k_kvb<<<kNB, 256, 0, stream>>>(keys, V, Ub, kvb);
    k_scan<<<kB, kThreads, SM_TOTAL, stream>>>(e, m, st0, keys, Vm, pa,
                                               UT, WT, kvb, (float*)d_out);
}
